// LoRAAttnProcessor_62148176773339
// MI455X (gfx1250) — compile-verified, matrix operand order checked by reading, not run on hardware
//
#include <hip/hip_runtime.h>
#include <stdint.h>

typedef __attribute__((ext_vector_type(16))) __bf16  v16bf;
typedef __attribute__((ext_vector_type(8)))  float   v8f;
typedef __attribute__((ext_vector_type(8)))  uint32_t uint8v;

#define WMMA_BF16(A, B, C) \
  __builtin_amdgcn_wmma_f32_16x16x32_bf16(false, (A), false, (B), (short)0, (C), false, false)

// Async fill: global -> LDS, 16B per lane, tracked by ASYNCcnt (no VGPR data path).
// vdst = wave-relative LDS byte address (low 32 bits of the flat shared pointer,
// per ISA 10.2 aperture rules), vaddr = 64-bit global address.
__device__ __forceinline__ void async_ld_b128(uint32_t lds_addr, const void* gptr) {
  asm volatile("global_load_async_to_lds_b128 %0, %1, off"
               :: "v"(lds_addr), "v"((unsigned long long)(uintptr_t)gptr)
               : "memory");
}
__device__ __forceinline__ uint32_t lds_off(const void* p) {
  return (uint32_t)(uintptr_t)p;
}

// Load a 16x32 bf16 fragment (A or B operand) per ISA 7.12.2:
// lane holds row (lane&15); VGPR i holds packed k-pair at
// k = (i>>2)*16 + (lane>>4)*8 + (i&3)*2  (+kofs for the chunk).
__device__ __forceinline__ v16bf frag_ld(const __bf16* base, int row, int stride,
                                         int kofs, int lane) {
  const __bf16* p = base + (row + (lane & 15)) * stride + kofs + ((lane >> 4) << 3);
  uint32_t w[8];
#pragma unroll
  for (int i = 0; i < 8; ++i) {
    int kp = ((i >> 2) << 4) + ((i & 3) << 1);
    w[i] = *(const uint32_t*)(p + kp);
  }
  uint8v u = {w[0], w[1], w[2], w[3], w[4], w[5], w[6], w[7]};
  return __builtin_bit_cast(v16bf, u);
}

// ---------------- elementwise prep ----------------

__global__ __launch_bounds__(256) void cvt_bf16_kernel(const float* __restrict__ x,
                                                       __bf16* __restrict__ y) {
  int i = (blockIdx.x * 256 + threadIdx.x) * 4;
  float4 f = *(const float4*)(x + i);
  y[i + 0] = (__bf16)f.x;
  y[i + 1] = (__bf16)f.y;
  y[i + 2] = (__bf16)f.z;
  y[i + 3] = (__bf16)f.w;
}

// W_eff[o,c] = W[o,c] + sum_r up[o,r]*down[r,c]   (rank-4 LoRA fold-in)
__global__ __launch_bounds__(256) void prep_w_kernel(const float* __restrict__ W,
                                                     const float* __restrict__ down,
                                                     const float* __restrict__ up,
                                                     __bf16* __restrict__ Weff) {
  int i = blockIdx.x * 256 + threadIdx.x;   // 0 .. 1024*1024-1
  int o = i >> 10, c = i & 1023;
  float acc = W[i];
#pragma unroll
  for (int r = 0; r < 4; ++r) acc += up[o * 4 + r] * down[r * 1024 + c];
  Weff[i] = (__bf16)acc;
}

// ---------------- shared GEMM core ----------------
// C[M,N=1024] tile 128x128 per WG; K staged 32-deep, double-buffered async-to-LDS.

struct GemmAcc {
  v8f acc[4][2];
};

__device__ __forceinline__ void gemm_core(const __bf16* __restrict__ Abase,
                                          const __bf16* __restrict__ Bbase,
                                          __bf16 (*la)[128][40], __bf16 (*lb)[128][40],
                                          int tid, int lane, int wm, int wn,
                                          GemmAcc& g) {
  v8f vzero = {0.f, 0.f, 0.f, 0.f, 0.f, 0.f, 0.f, 0.f};
#pragma unroll
  for (int mi = 0; mi < 4; ++mi)
#pragma unroll
    for (int ni = 0; ni < 2; ++ni) g.acc[mi][ni] = vzero;

  const int srow = tid >> 1, shalf = tid & 1;
  const __bf16* sa0 = Abase + srow * 1024 + shalf * 16;
  const __bf16* sb0 = Bbase + srow * 1024 + shalf * 16;

  // prologue: stage chunk 0 into buffer 0
  {
    async_ld_b128(lds_off(&la[0][srow][shalf * 16]),     sa0);
    async_ld_b128(lds_off(&la[0][srow][shalf * 16 + 8]), sa0 + 8);
    async_ld_b128(lds_off(&lb[0][srow][shalf * 16]),     sb0);
    async_ld_b128(lds_off(&lb[0][srow][shalf * 16 + 8]), sb0 + 8);
    asm volatile("s_wait_xcnt 0x0" ::: "memory");
  }

  for (int t = 0; t < 32; ++t) {
    if (t < 31) {  // stage chunk t+1 into the other buffer, then wait chunk t
      int nb = (t + 1) & 1, k0 = (t + 1) * 32;
      async_ld_b128(lds_off(&la[nb][srow][shalf * 16]),     sa0 + k0);
      async_ld_b128(lds_off(&la[nb][srow][shalf * 16 + 8]), sa0 + k0 + 8);
      async_ld_b128(lds_off(&lb[nb][srow][shalf * 16]),     sb0 + k0);
      async_ld_b128(lds_off(&lb[nb][srow][shalf * 16 + 8]), sb0 + k0 + 8);
      asm volatile("s_wait_xcnt 0x0" ::: "memory");
      asm volatile("s_wait_asynccnt 0x4" ::: "memory");
    } else {
      asm volatile("s_wait_asynccnt 0x0" ::: "memory");
    }
    __syncthreads();   // chunk t resident in LDS for all waves

    const int cb = t & 1;
    v16bf bf[2];
#pragma unroll
    for (int ni = 0; ni < 2; ++ni)
      bf[ni] = frag_ld(&lb[cb][0][0], wn * 32 + ni * 16, 40, 0, lane);
#pragma unroll
    for (int mi = 0; mi < 4; ++mi) {
      v16bf af = frag_ld(&la[cb][0][0], wm * 64 + mi * 16, 40, 0, lane);
#pragma unroll
      for (int ni = 0; ni < 2; ++ni)
        g.acc[mi][ni] = WMMA_BF16(af, bf[ni], g.acc[mi][ni]);
    }
    __syncthreads();   // all waves done reading buffer cb before it is re-filled
  }
}

// ---------------- QKV projection GEMM ----------------
// out -> [B,H,S,D] bf16; Q pre-scaled by 1/sqrt(64)

__global__ __launch_bounds__(256) void gemm_qkv_kernel(
    const __bf16* __restrict__ X, const __bf16* __restrict__ Wq,
    const __bf16* __restrict__ Wk, const __bf16* __restrict__ Wv,
    __bf16* __restrict__ Qo, __bf16* __restrict__ Ko, __bf16* __restrict__ Vo) {
  __shared__ __align__(16) __bf16 la[2][128][40];
  __shared__ __align__(16) __bf16 lb[2][128][40];
  const int tid = threadIdx.x, lane = tid & 31, wid = tid >> 5;
  const int wm = wid >> 2, wn = wid & 3;
  const int m0 = blockIdx.x * 128, n0 = blockIdx.y * 128;
  const int z = blockIdx.z;
  const __bf16* W = (z == 0) ? Wq : (z == 1) ? Wk : Wv;
  __bf16* D = (z == 0) ? Qo : (z == 1) ? Ko : Vo;
  const float scl = (z == 0) ? 0.125f : 1.0f;

  GemmAcc g;
  gemm_core(X + m0 * 1024, W + n0 * 1024, la, lb, tid, lane, wm, wn, g);

#pragma unroll
  for (int mi = 0; mi < 4; ++mi)
#pragma unroll
    for (int ni = 0; ni < 2; ++ni)
#pragma unroll
      for (int r = 0; r < 8; ++r) {
        int gm = m0 + wm * 64 + mi * 16 + ((lane >> 4) << 3) + r;
        int gc = n0 + wn * 32 + ni * 16 + (lane & 15);
        int b = gm >> 11, s = gm & 2047, h = gc >> 6, d = gc & 63;
        D[((((b << 4) + h) * 2048 + s) << 6) + d] = (__bf16)(g.acc[mi][ni][r] * scl);
      }
}

// ---------------- output projection GEMM (f32 out + bias) ----------------

__global__ __launch_bounds__(256) void gemm_o_kernel(
    const __bf16* __restrict__ A, const __bf16* __restrict__ W,
    const float* __restrict__ bias, float* __restrict__ out) {
  __shared__ __align__(16) __bf16 la[2][128][40];
  __shared__ __align__(16) __bf16 lb[2][128][40];
  const int tid = threadIdx.x, lane = tid & 31, wid = tid >> 5;
  const int wm = wid >> 2, wn = wid & 3;
  const int m0 = blockIdx.x * 128, n0 = blockIdx.y * 128;

  GemmAcc g;
  gemm_core(A + m0 * 1024, W + n0 * 1024, la, lb, tid, lane, wm, wn, g);

#pragma unroll
  for (int mi = 0; mi < 4; ++mi)
#pragma unroll
    for (int ni = 0; ni < 2; ++ni)
#pragma unroll
      for (int r = 0; r < 8; ++r) {
        int gm = m0 + wm * 64 + mi * 16 + ((lane >> 4) << 3) + r;
        int gc = n0 + wn * 32 + ni * 16 + (lane & 15);
        out[gm * 1024 + gc] = g.acc[mi][ni][r] + bias[gc];
      }
}

// ---------------- Flash attention ----------------
// grid: (S/128, B*H); 8 waves, each owns a 16-query strip; K/V tiles of 64.

__global__ __launch_bounds__(256) void attn_kernel(
    const __bf16* __restrict__ Q, const __bf16* __restrict__ K,
    const __bf16* __restrict__ V, __bf16* __restrict__ O) {
  __shared__ __align__(16) __bf16 kt[64][72];      // [key][d], async-filled
  __shared__ __align__(16) __bf16 vt[64][72];      // transposed: [d][key]
  __shared__ __align__(16) __bf16 pb[8][16][72];   // per-wave P re-layout buffer
  const int tid = threadIdx.x, lane = tid & 31, wid = tid >> 5;
  const int bh = blockIdx.y;
  const int qrow0 = blockIdx.x * 128 + wid * 16;
  const __bf16* Qb = Q + ((size_t)bh * 2048 + qrow0) * 64;
  const __bf16* Kb = K + (size_t)bh * 2048 * 64;
  const __bf16* Vb = V + (size_t)bh * 2048 * 64;

  v16bf qf[2];
#pragma unroll
  for (int c = 0; c < 2; ++c) qf[c] = frag_ld(Qb, 0, 64, c * 32, lane);

  v8f vzero = {0.f, 0.f, 0.f, 0.f, 0.f, 0.f, 0.f, 0.f};
  v8f oacc[4];
#pragma unroll
  for (int dt = 0; dt < 4; ++dt) oacc[dt] = vzero;
  float mrow[8], lrow[8];
#pragma unroll
  for (int r = 0; r < 8; ++r) { mrow[r] = -3.0e38f; lrow[r] = 0.f; }

  const int srow = tid >> 2, sseg = tid & 3;
  for (int t0 = 0; t0 < 2048; t0 += 64) {
    {   // stage K tile via async-to-LDS (no VGPR round trip)
      const __bf16* src = Kb + (t0 + srow) * 64 + sseg * 16;
      async_ld_b128(lds_off(&kt[srow][sseg * 16]),     src);
      async_ld_b128(lds_off(&kt[srow][sseg * 16 + 8]), src + 8);
    }
    {   // stage V tile transposed (software transpose through VGPRs)
      const __bf16* src = Vb + (t0 + srow) * 64 + sseg * 16;
      __align__(16) __bf16 tmp[16];
      *(uint4*)&tmp[0] = *(const uint4*)src;
      *(uint4*)&tmp[8] = *(const uint4*)(src + 8);
#pragma unroll
      for (int j = 0; j < 16; ++j) vt[sseg * 16 + j][srow] = tmp[j];
    }
    asm volatile("s_wait_xcnt 0x0" ::: "memory");
    asm volatile("s_wait_asynccnt 0x0" ::: "memory");
    __syncthreads();

    // scores S = Q(16x64) * K_tile^T  -> 4 tiles of 16x16
    v8f sc[4];
#pragma unroll
    for (int ni = 0; ni < 4; ++ni) {
      v8f s = vzero;
      v16bf b0 = frag_ld(&kt[0][0], ni * 16, 72, 0, lane);
      v16bf b1 = frag_ld(&kt[0][0], ni * 16, 72, 32, lane);
      s = WMMA_BF16(qf[0], b0, s);
      s = WMMA_BF16(qf[1], b1, s);
      sc[ni] = s;
    }

    // online softmax update (rows M=(lane>>4)*8+r, cols lane&15)
#pragma unroll
    for (int r = 0; r < 8; ++r) {
      float mv = fmaxf(fmaxf(sc[0][r], sc[1][r]), fmaxf(sc[2][r], sc[3][r]));
#pragma unroll
      for (int off = 1; off < 16; off <<= 1) mv = fmaxf(mv, __shfl_xor(mv, off, 32));
      float mcur = fmaxf(mrow[r], mv);
      float al = __expf(mrow[r] - mcur);
      mrow[r] = mcur;
      lrow[r] *= al;
#pragma unroll
      for (int dt = 0; dt < 4; ++dt) oacc[dt][r] *= al;
      float ps = 0.f;
      int M = ((lane >> 4) << 3) + r;
#pragma unroll
      for (int ni = 0; ni < 4; ++ni) {
        float pv = __expf(sc[ni][r] - mcur);
        ps += pv;
        pb[wid][M][ni * 16 + (lane & 15)] = (__bf16)pv;
      }
#pragma unroll
      for (int off = 1; off < 16; off <<= 1) ps += __shfl_xor(ps, off, 32);
      lrow[r] += ps;
    }

    asm volatile("s_wait_dscnt 0x0" ::: "memory");   // wave-local P write->read

    v16bf pf0 = frag_ld(&pb[wid][0][0], 0, 72, 0, lane);
    v16bf pf1 = frag_ld(&pb[wid][0][0], 0, 72, 32, lane);
#pragma unroll
    for (int dt = 0; dt < 4; ++dt) {
      v16bf v0 = frag_ld(&vt[0][0], dt * 16, 72, 0, lane);
      v16bf v1 = frag_ld(&vt[0][0], dt * 16, 72, 32, lane);
      oacc[dt] = WMMA_BF16(pf0, v0, oacc[dt]);
      oacc[dt] = WMMA_BF16(pf1, v1, oacc[dt]);
    }
    __syncthreads();
  }

  const int b = bh >> 4, h = bh & 15;
#pragma unroll
  for (int r = 0; r < 8; ++r) {
    float inv = 1.0f / lrow[r];
    int s = qrow0 + ((lane >> 4) << 3) + r;
#pragma unroll
    for (int dt = 0; dt < 4; ++dt) {
      int c = h * 64 + dt * 16 + (lane & 15);
      O[((size_t)(b * 2048 + s) << 10) + c] = (__bf16)(oacc[dt][r] * inv);
    }
  }
}

// ---------------- launch ----------------

extern "C" void kernel_launch(void* const* d_in, const int* in_sizes, int n_in,
                              void* d_out, int out_size, void* d_ws, size_t ws_size,
                              hipStream_t stream) {
  const float* hs = (const float*)d_in[0];
  const float* Wq = (const float*)d_in[1];
  const float* Wk = (const float*)d_in[2];
  const float* Wv = (const float*)d_in[3];
  const float* Wo = (const float*)d_in[4];
  const float* bo = (const float*)d_in[5];
  const float* qd = (const float*)d_in[6];
  const float* qu = (const float*)d_in[7];
  const float* kd = (const float*)d_in[8];
  const float* ku = (const float*)d_in[9];
  const float* vd = (const float*)d_in[10];
  const float* vu = (const float*)d_in[11];
  const float* od = (const float*)d_in[12];
  const float* ou = (const float*)d_in[13];

  const size_t MB = 1024ull * 1024ull;
  char* ws = (char*)d_ws;
  __bf16* Xb  = (__bf16*)(ws + 0 * MB);    // 8192x1024 bf16  (16 MB)
  __bf16* Wqe = (__bf16*)(ws + 16 * MB);   // 1024x1024 bf16  (2 MB each)
  __bf16* Wke = (__bf16*)(ws + 18 * MB);
  __bf16* Wve = (__bf16*)(ws + 20 * MB);
  __bf16* Woe = (__bf16*)(ws + 22 * MB);
  __bf16* Qs  = (__bf16*)(ws + 24 * MB);   // [B,H,S,D] bf16  (16 MB each)
  __bf16* Ks  = (__bf16*)(ws + 40 * MB);
  __bf16* Vs  = (__bf16*)(ws + 56 * MB);
  __bf16* Ao  = (__bf16*)(ws + 72 * MB);   // [B,S,C] bf16    (16 MB)
  float* out = (float*)d_out;

  cvt_bf16_kernel<<<8192, 256, 0, stream>>>(hs, Xb);
  prep_w_kernel<<<4096, 256, 0, stream>>>(Wq, qd, qu, Wqe);
  prep_w_kernel<<<4096, 256, 0, stream>>>(Wk, kd, ku, Wke);
  prep_w_kernel<<<4096, 256, 0, stream>>>(Wv, vd, vu, Wve);
  prep_w_kernel<<<4096, 256, 0, stream>>>(Wo, od, ou, Woe);

  gemm_qkv_kernel<<<dim3(64, 8, 3), 256, 0, stream>>>(Xb, Wqe, Wke, Wve, Qs, Ks, Vs);
  attn_kernel<<<dim3(16, 64), 256, 0, stream>>>(Qs, Ks, Vs, Ao);
  gemm_o_kernel<<<dim3(64, 8), 256, 0, stream>>>(Ao, Woe, bo, out);
}
